// BertMHSelfAttention_72825465471245
// MI455X (gfx1250) — matrix operand order repaired and checked by reading, not compiled, not run on hardware
//
#include <hip/hip_runtime.h>
#include <hip/hip_bf16.h>

#define BDIM 256
constexpr int B_ = 16, T_ = 8192, C_ = 1024, H_ = 16, D_ = 64;
constexpr int NCH = 32;          // chunks per batch (blocks along T)
constexpr int SPAN = T_ / NCH;   // 256 rows per block
constexpr int TS = 32;           // subtile rows processed per iteration
constexpr int NSUB = SPAN / TS;  // 8

typedef __attribute__((ext_vector_type(16))) __bf16 v16bf;
typedef __attribute__((ext_vector_type(8)))  float  v8f;

union ABf { v16bf v; unsigned int w[8]; };

__device__ __forceinline__ float b2f(unsigned short u) {
  return __uint_as_float(((unsigned int)u) << 16);
}
__device__ __forceinline__ unsigned short f2b(float f) {
  unsigned int x = __float_as_uint(f);
  x += 0x7fffu + ((x >> 16) & 1u);   // round-to-nearest-even
  return (unsigned short)(x >> 16);
}

// ---------------------------------------------------------------------------
// Kernel 1: q = (x[b,0,:]+pe0) @ Wq + bq ;  u[b,h,:] = scale * Wk_h @ q_h (bf16)
//           cbias[b,h] = scale * bk_h . q_h
// ---------------------------------------------------------------------------
__global__ void __launch_bounds__(BDIM)
k1_prep(const float* __restrict__ x, const float* __restrict__ Wq,
        const float* __restrict__ bq, const float* __restrict__ Wkv,
        const float* __restrict__ bkv, unsigned short* __restrict__ ut,
        float* __restrict__ cb) {
  __shared__ float xq[C_];
  __shared__ float qsh[C_];
  const int b = blockIdx.x, tid = threadIdx.x;
  // pe at position 0: even dims sin(0)=0, odd dims cos(0)=1
  for (int c = tid; c < C_; c += BDIM)
    xq[c] = x[(size_t)b * T_ * C_ + c] + ((c & 1) ? 1.0f : 0.0f);
  __syncthreads();
  for (int j = tid; j < C_; j += BDIM) {
    float acc = bq[j];
    for (int c = 0; c < C_; ++c) acc += xq[c] * Wq[(size_t)c * C_ + j];
    qsh[j] = acc;
  }
  __syncthreads();
  const float scale = 0.125f;  // 1/sqrt(D)
  for (int idx = tid; idx < H_ * C_; idx += BDIM) {
    int h = idx >> 10, c = idx & (C_ - 1);
    const float* wk = Wkv + (size_t)c * (2 * C_) + h * D_;  // K half of Wkv
    float acc = 0.f;
#pragma unroll 4
    for (int d = 0; d < D_; ++d) acc += qsh[h * D_ + d] * wk[d];
    ut[((size_t)b * H_ + h) * C_ + c] = f2b(acc * scale);
  }
  if (tid < H_) {
    float acc = 0.f;
    for (int d = 0; d < D_; ++d) acc += bkv[tid * D_ + d] * qsh[tid * D_ + d];
    cb[b * H_ + tid] = acc * scale;
  }
}

// ---------------------------------------------------------------------------
// Kernel 2: single-pass flash decode over x (+PE on the fly), all-bf16 WMMA
// ---------------------------------------------------------------------------
constexpr int XB_S = 1032;   // bf16 row stride: 516 dwords == 4 mod 64 (conflict-free)
constexpr int XT_S = 40;     // bf16 col-major stride: 20 dwords mod 64 (conflict-free)
constexpr int UT_S = 1032;
constexpr int SP_S = 36;     // padded t-stride for f32 score buffers (36 mod 64)
constexpr int PB_S = 40;     // bf16 p stride

constexpr int XB_OFF  = 0;                        // ushort [32][1032] row-major x tile
constexpr int XT_OFF  = XB_OFF + TS * XB_S * 2;   // ushort [1024][40] col-major x tile
constexpr int UT_OFF  = XT_OFF + C_ * XT_S * 2;   // ushort [16][1032] U^T
constexpr int SP_OFF  = UT_OFF + 16 * UT_S * 2;   // float  [8][16][36] K-split partials
constexpr int ST_OFF  = SP_OFF + 8 * 16 * SP_S * 4;
constexpr int PB_OFF  = ST_OFF + 16 * SP_S * 4;   // ushort [16][40] bf16 softmax weights
constexpr int DT_OFF  = PB_OFF + 16 * PB_S * 2;   // float [512] div_term
constexpr int RED_OFF = DT_OFF + 512 * 4;         // float [16][16]
constexpr int CBL_OFF = RED_OFF + 256 * 4;
constexpr int MM_OFF  = CBL_OFF + 64;
constexpr int LL_OFF  = MM_OFF + 64;
constexpr int FC_OFF  = LL_OFF + 64;
constexpr int K2_LDS  = FC_OFF + 64;              // ~206 KB of 320 KB WGP LDS

__global__ void __launch_bounds__(BDIM)
k2_flash(const float* __restrict__ x, const unsigned short* __restrict__ ut,
         const float* __restrict__ cb, float* __restrict__ mg,
         float* __restrict__ lg, float* __restrict__ zg) {
  extern __shared__ char smem[];
  unsigned short* xbL = (unsigned short*)(smem + XB_OFF);
  unsigned short* xtL = (unsigned short*)(smem + XT_OFF);
  unsigned short* utL = (unsigned short*)(smem + UT_OFF);
  float* spL  = (float*)(smem + SP_OFF);
  float* stL  = (float*)(smem + ST_OFF);
  unsigned short* pbL = (unsigned short*)(smem + PB_OFF);
  float* dtL  = (float*)(smem + DT_OFF);
  float* redL = (float*)(smem + RED_OFF);
  float* cbL  = (float*)(smem + CBL_OFF);
  float* mLc  = (float*)(smem + MM_OFF);
  float* lLc  = (float*)(smem + LL_OFF);
  float* fcL  = (float*)(smem + FC_OFF);

  const int tid = threadIdx.x;
  const int wave = tid >> 5, lane = tid & 31;
  const int hi = (lane >> 4) & 1, l15 = lane & 15;
  const int chunk = blockIdx.x, b = blockIdx.y;
  const int t0 = chunk * SPAN;

  // div_term[i] = exp(-ln(10000)*2i/C)
  for (int i = tid; i < 512; i += BDIM)
    dtL[i] = __expf(-0.017988946f * (float)i);
  // stage U^T[b] (bf16) into LDS once per block
  for (int i = tid; i < H_ * C_; i += BDIM) {
    int h = i >> 10, c = i & (C_ - 1);
    utL[h * UT_S + c] = ut[((size_t)b * H_ + h) * C_ + c];
  }
  if (tid < H_) {
    cbL[tid] = cb[b * H_ + tid];
    mLc[tid] = -1e30f;
    lLc[tid] = 0.f;
  }
  v8f zacc[8];
#pragma unroll
  for (int ct = 0; ct < 8; ++ct)
#pragma unroll
    for (int e = 0; e < 8; ++e) zacc[ct][e] = 0.f;
  __syncthreads();

  const float4* xg = (const float4*)(x + ((size_t)b * T_ + t0) * C_);
  constexpr int NIT = (TS * C_ / 4) / BDIM;   // 32 float4 iters per subtile

  for (int it = 0; it < NSUB; ++it) {
    // ---- stage TS rows of x: +PE on the fly, bf16, BOTH layouts ----
    const float4* xs = xg + (size_t)it * (TS * C_ / 4);
    for (int k = 0; k < NIT; ++k) {
      int idx4 = k * BDIM + tid;
      int row = idx4 >> 8;                  // 256 float4 per row
      int c4 = (idx4 & 255) << 2;
      float4 xv = xs[idx4];
      if (it + 1 < NSUB)                    // global_prefetch_b8 next subtile
        __builtin_prefetch((const char*)&xs[idx4 + TS * C_ / 4], 0, 1);
      float tpos = (float)(t0 + it * TS + row);
      float a0 = tpos * dtL[c4 >> 1];
      float a1 = tpos * dtL[(c4 >> 1) + 1];
      xv.x += __sinf(a0); xv.y += __cosf(a0);
      xv.z += __sinf(a1); xv.w += __cosf(a1);
      unsigned short e0 = f2b(xv.x), e1 = f2b(xv.y), e2 = f2b(xv.z), e3 = f2b(xv.w);
      unsigned int* dst = (unsigned int*)&xbL[row * XB_S + c4];
      dst[0] = (unsigned int)e0 | ((unsigned int)e1 << 16);
      dst[1] = (unsigned int)e2 | ((unsigned int)e3 << 16);
      xtL[(c4 + 0) * XT_S + row] = e0;      // transposed copy for the z GEMM
      xtL[(c4 + 1) * XT_S + row] = e1;
      xtL[(c4 + 2) * XT_S + row] = e2;
      xtL[(c4 + 3) * XT_S + row] = e3;
    }
    __syncthreads();

    // ---- score GEMM: S^T(16h x 32t) = U^T(16x1024) @ Xc^T, K split 8 ways ----
    {
      const int c0w = wave * 128;
#pragma unroll
      for (int nt = 0; nt < 2; ++nt) {
        v8f acc;
#pragma unroll
        for (int e = 0; e < 8; ++e) acc[e] = 0.f;
        const int tloc = nt * 16 + l15;
#pragma unroll
        for (int ks = 0; ks < 4; ++ks) {
          const int cbase = c0w + ks * 32;
          ABf A, Bf;
#pragma unroll
          for (int v = 0; v < 8; ++v) {
            int ka = ((v < 4) ? 2 * v : 16 + 2 * (v - 4)) + (hi ? 8 : 0);
            A.w[v] = *(const unsigned int*)&utL[l15 * UT_S + cbase + ka];
            int kb = ((v < 4) ? 2 * v : 16 + 2 * (v - 4)) + (hi ? 8 : 0);
            Bf.w[v] = *(const unsigned int*)&xbL[tloc * XB_S + cbase + kb];
          }
          acc = __builtin_amdgcn_wmma_f32_16x16x32_bf16(
              false, A.v, false, Bf.v, (short)0, acc, false, false);
        }
#pragma unroll
        for (int e = 0; e < 8; ++e) {
          int h = e + (hi ? 8 : 0);
          spL[(wave * 16 + h) * SP_S + tloc] = acc[e];
        }
      }
    }
    __syncthreads();

    // ---- online softmax: reduce K-split partials, update (m, l), build P ----
    {
      int h = tid & 15, g = tid >> 4;
      float mx = -1e30f;
#pragma unroll
      for (int tt = 0; tt < 2; ++tt) {
        int t = g * 2 + tt;
        float tot = cbL[h];
#pragma unroll
        for (int w = 0; w < 8; ++w) tot += spL[(w * 16 + h) * SP_S + t];
        stL[h * SP_S + t] = tot;
        mx = fmaxf(mx, tot);
      }
      redL[g * 16 + h] = mx;
    }
    __syncthreads();
    if (tid < H_) {
      float mt = -1e30f;
      for (int g = 0; g < 16; ++g) mt = fmaxf(mt, redL[g * 16 + tid]);
      float mnew = fmaxf(mLc[tid], mt);
      fcL[tid] = __expf(mLc[tid] - mnew);
      mLc[tid] = mnew;
    }
    __syncthreads();
    {
      int h = tid & 15, g = tid >> 4;
      float s = 0.f;
#pragma unroll
      for (int tt = 0; tt < 2; ++tt) {
        int t = g * 2 + tt;
        float p = __expf(stL[h * SP_S + t] - mLc[h]);
        unsigned short pb = f2b(p);
        pbL[h * PB_S + t] = pb;
        s += b2f(pb);          // denominator uses the SAME rounded weight
      }
      redL[g * 16 + h] = s;
    }
    __syncthreads();
    if (tid < H_) {
      float s = 0.f;
      for (int g = 0; g < 16; ++g) s += redL[g * 16 + tid];
      lLc[tid] = lLc[tid] * fcL[tid] + s;
    }

    // ---- rescale z accumulators by exp(m_old - m_new) per head row ----
    float fs[8];
#pragma unroll
    for (int e = 0; e < 8; ++e) fs[e] = fcL[e + (hi ? 8 : 0)];
#pragma unroll
    for (int ct = 0; ct < 8; ++ct)
#pragma unroll
      for (int e = 0; e < 8; ++e) zacc[ct][e] *= fs[e];

    // ---- z GEMM (bf16 WMMA): Z(16h x 1024c) += P^T(16x32) @ Xc(32x1024) ----
    {
      const int colw = wave * 128;   // each wave owns a 128-column slice
      ABf A;
#pragma unroll
      for (int v = 0; v < 8; ++v) {
        int ka = ((v < 4) ? 2 * v : 16 + 2 * (v - 4)) + (hi ? 8 : 0);
        A.w[v] = *(const unsigned int*)&pbL[l15 * PB_S + ka];
      }
#pragma unroll
      for (int ct = 0; ct < 8; ++ct) {
        const int col = colw + ct * 16 + l15;
        ABf Bf;
#pragma unroll
        for (int v = 0; v < 8; ++v) {
          int kb = ((v < 4) ? 2 * v : 16 + 2 * (v - 4)) + (hi ? 8 : 0);
          Bf.w[v] = *(const unsigned int*)&xtL[col * XT_S + kb];
        }
        zacc[ct] = __builtin_amdgcn_wmma_f32_16x16x32_bf16(
            false, A.v, false, Bf.v, (short)0, zacc[ct], false, false);
      }
    }
    __syncthreads();
  }

  // ---- write block partials (m, l, z) ----
#pragma unroll
  for (int ct = 0; ct < 8; ++ct)
#pragma unroll
    for (int e = 0; e < 8; ++e) {
      int h = e + (hi ? 8 : 0);
      int col = wave * 128 + ct * 16 + l15;
      zg[(((size_t)b * NCH + chunk) * H_ + h) * C_ + col] = zacc[ct][e];
    }
  if (tid < H_) {
    mg[(b * NCH + chunk) * H_ + tid] = mLc[tid];
    lg[(b * NCH + chunk) * H_ + tid] = lLc[tid];
  }
}

// ---------------------------------------------------------------------------
// Kernel 3: merge chunk partials, apply V projection + output projection
// ---------------------------------------------------------------------------
constexpr int K3_E_OFF  = 0;                 // float [32][16]
constexpr int K3_LH_OFF = 2048;              // float [16]
constexpr int K3_YA_OFF = 2112;              // float [1024]
constexpr int K3_ZN_OFF = 2112 + 4096;       // float [16*1024]
constexpr int K3_LDS    = K3_ZN_OFF + 65536;

__global__ void __launch_bounds__(BDIM)
k3_out(const float* __restrict__ Wkv, const float* __restrict__ bkv,
       const float* __restrict__ Wo, const float* __restrict__ bo,
       const float* __restrict__ mg, const float* __restrict__ lg,
       const float* __restrict__ zg, float* __restrict__ out) {
  extern __shared__ char smem[];
  float* E  = (float*)(smem + K3_E_OFF);
  float* Lh = (float*)(smem + K3_LH_OFF);
  float* ya = (float*)(smem + K3_YA_OFF);
  float* zn = (float*)(smem + K3_ZN_OFF);
  const int b = blockIdx.x, tid = threadIdx.x;
  if (tid < H_) {
    float M = -1e30f;
    for (int j = 0; j < NCH; ++j) M = fmaxf(M, mg[(b * NCH + j) * H_ + tid]);
    float L = 0.f;
    for (int j = 0; j < NCH; ++j) {
      float e = __expf(mg[(b * NCH + j) * H_ + tid] - M);
      E[j * 16 + tid] = e;
      L += e * lg[(b * NCH + j) * H_ + tid];
    }
    Lh[tid] = L;
  }
  __syncthreads();
  for (int idx = tid; idx < H_ * C_; idx += BDIM) {
    int h = idx >> 10, c = idx & (C_ - 1);
    float acc = 0.f;
    for (int j = 0; j < NCH; ++j)
      acc += E[j * 16 + h] * zg[(((size_t)b * NCH + j) * H_ + h) * C_ + c];
    zn[idx] = acc / Lh[h];
  }
  __syncthreads();
  // y_attn[h*64+d] = zn[h,:] . Wv[:, h*64+d] + bv   (V half of Wkv)
  for (int i = tid; i < C_; i += BDIM) {
    int h = i >> 6;
    float acc = bkv[C_ + i];
    const float* zrow = zn + h * C_;
    for (int c = 0; c < C_; ++c)
      acc += zrow[c] * Wkv[(size_t)c * (2 * C_) + C_ + i];
    ya[i] = acc;
  }
  __syncthreads();
  for (int j = tid; j < C_; j += BDIM) {
    float acc = bo[j];
    for (int i = 0; i < C_; ++i) acc += ya[i] * Wo[(size_t)i * C_ + j];
    out[(size_t)b * C_ + j] = acc;
  }
}

// ---------------------------------------------------------------------------
extern "C" void kernel_launch(void* const* d_in, const int* in_sizes, int n_in,
                              void* d_out, int out_size, void* d_ws, size_t ws_size,
                              hipStream_t stream) {
  const float* x   = (const float*)d_in[0];
  const float* Wq  = (const float*)d_in[1];
  const float* bq  = (const float*)d_in[2];
  const float* Wkv = (const float*)d_in[3];
  const float* bkv = (const float*)d_in[4];
  const float* Wo  = (const float*)d_in[5];
  const float* bo  = (const float*)d_in[6];
  float* out = (float*)d_out;

  char* ws = (char*)d_ws;
  unsigned short* ut = (unsigned short*)ws;                    // 512 KB bf16 U^T
  float* cb = (float*)(ws + 524288);                           // 1 KB (pad 4 KB)
  float* mg = (float*)(ws + 524288 + 4096);                    // 32 KB
  float* lg = (float*)(ws + 524288 + 4096 + 32768);            // 32 KB
  float* zg = (float*)(ws + 524288 + 4096 + 65536);            // 32 MB partial z

  k1_prep<<<B_, BDIM, 0, stream>>>(x, Wq, bq, Wkv, bkv, ut, cb);
  k2_flash<<<dim3(NCH, B_), BDIM, K2_LDS, stream>>>(x, ut, cb, mg, lg, zg);
  k3_out<<<B_, BDIM, K3_LDS, stream>>>(Wkv, bkv, Wo, bo, mg, lg, zg, out);
}
